// MoE_31696858645001
// MI455X (gfx1250) — matrix operand order repaired and checked by reading, not compiled, not run on hardware
//
#include <hip/hip_runtime.h>
#include <hip/hip_bf16.h>

#define D_MODEL 512
#define D_FF    2048
#define NUM_E   8
#define FH      (D_FF / 2)          // F handled per pass = 1024

typedef __bf16 v16bf __attribute__((ext_vector_type(16)));
typedef __bf16 v8bf  __attribute__((ext_vector_type(8)));
typedef float  v8f   __attribute__((ext_vector_type(8)));

__device__ __forceinline__ unsigned short f32_to_bf16(float f) {
  unsigned int u = __builtin_bit_cast(unsigned int, f);
  u += 0x7fffu + ((u >> 16) & 1u);          // round-to-nearest-even
  return (unsigned short)(u >> 16);
}

__device__ __forceinline__ v8f splat8(float f) {
  v8f v;
#pragma unroll
  for (int i = 0; i < 8; ++i) v[i] = f;
  return v;
}

// Build a v16bf fragment from two 16-byte chunks of bf16 data.
__device__ __forceinline__ v16bf load_frag(const unsigned short* p0,
                                           const unsigned short* p1) {
  v16bf a;
  ((v8bf*)&a)[0] = *(const v8bf*)p0;
  ((v8bf*)&a)[1] = *(const v8bf*)p1;
  return a;
}

// ---------------------------------------------------------------------------
// Kernel 1: gating  (logits -> top-2 -> renormalized gates + expert counts)
// ---------------------------------------------------------------------------
__global__ __launch_bounds__(256) void moe_gate(
    const float* __restrict__ x, const float* __restrict__ Wg,
    const float* __restrict__ bg, int T,
    int* __restrict__ topi, float* __restrict__ topv, int* __restrict__ counts)
{
  __shared__ float WgS[D_MODEL * NUM_E];   // 16 KB
  __shared__ float bgS[NUM_E];
  for (int i = threadIdx.x; i < D_MODEL * NUM_E; i += 256) WgS[i] = Wg[i];
  if (threadIdx.x < NUM_E) bgS[threadIdx.x] = bg[threadIdx.x];
  __syncthreads();

  int t = blockIdx.x * 256 + threadIdx.x;
  if (t >= T) return;

  float acc[NUM_E];
#pragma unroll
  for (int e = 0; e < NUM_E; ++e) acc[e] = bgS[e];

  const float4* x4 = (const float4*)(x + (size_t)t * D_MODEL);
  for (int d4 = 0; d4 < D_MODEL / 4; ++d4) {
    float4 v = x4[d4];
    int b0 = d4 * 4 * NUM_E;
#pragma unroll
    for (int e = 0; e < NUM_E; ++e) {
      acc[e] += v.x * WgS[b0 + e]      + v.y * WgS[b0 + 8 + e] +
                v.z * WgS[b0 + 16 + e] + v.w * WgS[b0 + 24 + e];
    }
  }

  float m = acc[0];
#pragma unroll
  for (int e = 1; e < NUM_E; ++e) m = fmaxf(m, acc[e]);
  float p[NUM_E];
#pragma unroll
  for (int e = 0; e < NUM_E; ++e) p[e] = __expf(acc[e] - m);
  // softmax denominator cancels in v0/(v0+v1)

  int i0 = 0; float v0 = p[0];
#pragma unroll
  for (int e = 1; e < NUM_E; ++e) if (p[e] > v0) { v0 = p[e]; i0 = e; }
  int i1 = -1; float v1 = -1.f;
#pragma unroll
  for (int e = 0; e < NUM_E; ++e) if (e != i0 && p[e] > v1) { v1 = p[e]; i1 = e; }

  float r = 1.f / (v0 + v1);
  topi[2 * t] = i0;  topi[2 * t + 1] = i1;
  topv[2 * t] = v0 * r; topv[2 * t + 1] = v1 * r;
  atomicAdd(&counts[i0], 1);
  atomicAdd(&counts[i1], 1);
}

// ---------------------------------------------------------------------------
// Kernel 2: tiny scans.  meta layout:
//   [0..7] counts  [8..16] token offsets  [17..25] 32-row tile offsets
//   [26..33] write pointers
// ---------------------------------------------------------------------------
__global__ void moe_scan(int* __restrict__ meta)
{
  if (threadIdx.x == 0 && blockIdx.x == 0) {
    int off = 0, toff = 0;
    meta[8] = 0; meta[17] = 0;
    for (int e = 0; e < NUM_E; ++e) {
      int c = meta[e];
      meta[26 + e] = off;
      off += c;                meta[8 + e + 1]  = off;
      toff += (c + 31) >> 5;   meta[17 + e + 1] = toff;
    }
  }
}

// ---------------------------------------------------------------------------
// Kernel 3: scatter tokens into expert-sorted lists
// ---------------------------------------------------------------------------
__global__ __launch_bounds__(256) void moe_scatter(
    int T, const int* __restrict__ topi, const float* __restrict__ topv,
    int* __restrict__ wptr, int* __restrict__ tok_of, float* __restrict__ gate_of)
{
  int t = blockIdx.x * 256 + threadIdx.x;
  if (t >= T) return;
#pragma unroll
  for (int k = 0; k < 2; ++k) {
    int e = topi[2 * t + k];
    int pos = atomicAdd(&wptr[e], 1);
    tok_of[pos]  = t;
    gate_of[pos] = topv[2 * t + k];
  }
}

// ---------------------------------------------------------------------------
// Kernel 4: convert f32 weights [E][K][N] -> bf16 in WMMA B-fragment layout:
//   dst linear = (((e*NT + nt)*KB + kb)*32 + lane)*16 + j
//   lane<16 : col = nt*16+lane, K = kb*32 + j
//   lane>=16: col = nt*16+(lane-16), K = kb*32 + 16 + j
// Each lane's 16 bf16 are contiguous (32 B) -> two b128 loads at GEMM time.
// ---------------------------------------------------------------------------
__global__ __launch_bounds__(256) void moe_wconv(
    const float* __restrict__ src, unsigned short* __restrict__ dst,
    int K, int N, long total)
{
  long idx = (long)blockIdx.x * 256 + threadIdx.x;
  if (idx >= total) return;
  int  KB  = K >> 5;
  long pes = (long)K * N;
  int  e   = (int)(idx / pes);
  long r   = idx - (long)e * pes;
  int  j    = (int)(r & 15);
  int  lane = (int)((r >> 4) & 31);
  int  kb   = (int)((r >> 9) % KB);
  int  nt   = (int)(r / (512 * (long)KB));
  int  k    = kb * 32 + ((lane & 16) ? 16 : 0) + j;
  int  n    = nt * 16 + (lane & 15);
  dst[idx] = f32_to_bf16(src[(long)e * pes + (long)k * N + n]);
}

// ---------------------------------------------------------------------------
// Kernel 5: fused expert FFN over 32-token tiles (bf16 WMMA, f32 accumulate).
// Two passes over the F dimension (1024 each) so Hs fits comfortably in LDS;
// each B fragment feeds two WMMAs (two 16-row M-subtiles) -> half the
// B-operand traffic per FLOP vs a 16-token tile.
// ---------------------------------------------------------------------------
#define XS_STRIDE 520     // 512 + 8 bf16 pad (16B-aligned rows, conflict-free)
#define HS_STRIDE 1032    // 1024 + 8

__global__ __launch_bounds__(256) void moe_ffn(
    const float* __restrict__ x,  const float* __restrict__ b1,
    const float* __restrict__ b2,
    const unsigned short* __restrict__ W1f, const unsigned short* __restrict__ W2f,
    const int* __restrict__ meta, const int* __restrict__ tok_of,
    const float* __restrict__ gate_of, float* __restrict__ out)
{
  __shared__ __align__(16) unsigned short Xs[32 * XS_STRIDE]; // 33.3 KB
  __shared__ __align__(16) unsigned short Hs[32 * HS_STRIDE]; // 66.0 KB
  __shared__ int   tokS[32];
  __shared__ float gateS[32];

  int b = blockIdx.x;
  if (b >= meta[17 + NUM_E]) return;            // past real tile count

  int e = 0;
  while (e < NUM_E - 1 && b >= meta[17 + e + 1]) ++e;
  int rowStart = meta[8 + e] + (b - meta[17 + e]) * 32;
  int nvalid   = meta[8 + e + 1] - rowStart;
  nvalid = nvalid > 32 ? 32 : nvalid;

  int tid = threadIdx.x;
  if (tid < 32) {
    int tok = -1; float g = 0.f;
    if (tid < nvalid) { tok = tok_of[rowStart + tid]; g = gate_of[rowStart + tid]; }
    tokS[tid] = tok; gateS[tid] = g;
  }
  __syncthreads();

  { // gather + convert X tile: 32 rows x 512 f32 -> bf16 LDS
    int row = tid & 31;
    int c0  = (tid >> 5) * 64;                  // 8 threads-per-row * 64 cols
    int tok = tokS[row];
    const float* xr = x + (size_t)(tok < 0 ? 0 : tok) * D_MODEL;
#pragma unroll
    for (int c = 0; c < 64; c += 4) {
      float4 v = make_float4(0.f, 0.f, 0.f, 0.f);
      if (tok >= 0) v = *(const float4*)(xr + c0 + c);
      Xs[row * XS_STRIDE + c0 + c + 0] = f32_to_bf16(v.x);
      Xs[row * XS_STRIDE + c0 + c + 1] = f32_to_bf16(v.y);
      Xs[row * XS_STRIDE + c0 + c + 2] = f32_to_bf16(v.z);
      Xs[row * XS_STRIDE + c0 + c + 3] = f32_to_bf16(v.w);
    }
  }

  int wave  = tid >> 5;
  int lane  = tid & 31;
  int lrow  = lane & 15;
  int khalf = (lane >> 4) << 4;      // A-frag K sub-offset: 0 or 16
  int rowb  = (lane >> 4) * 8;       // C/D row base within a 16-row subtile

  const unsigned short* W1e = W1f + (size_t)e * (D_MODEL * D_FF);
  const unsigned short* W2e = W2f + (size_t)e * (D_FF * D_MODEL);
  const float* b1e = b1 + e * D_FF;
  const float* b2e = b2 + e * D_MODEL;

  // persistent phase-2 accumulators: 4 N-tiles x 2 M-subtiles (bias preloaded)
  v8f oacc[4][2];
#pragma unroll
  for (int i = 0; i < 4; ++i) {
    float bias = b2e[(wave + 8 * i) * 16 + lrow];
    oacc[i][0] = splat8(bias);
    oacc[i][1] = splat8(0.f);        // bias only once per output element
  }
  // note: subtile m=1 contributes to DIFFERENT rows, so it needs the bias too
#pragma unroll
  for (int i = 0; i < 4; ++i) oacc[i][1] = oacc[i][0];

  for (int p = 0; p < 2; ++p) {                  // two halves of the F dim
    __syncthreads();                             // Hs safe to overwrite

    // ---- phase 1: Hs[32, FH] = relu(X @ W1[:, p*FH : (p+1)*FH] + b1) ----
    for (int ntl = wave; ntl < FH / 16; ntl += 8) {
      int nt = p * (FH / 16) + ntl;              // global F-tile index
      float bias = b1e[nt * 16 + lrow];
      v8f c0a = splat8(bias);
      v8f c1a = splat8(bias);
      const unsigned short* bp =
          W1e + ((size_t)nt * (D_MODEL / 32)) * 512 + lane * 16;
      __builtin_prefetch(bp, 0, 1);
#pragma unroll
      for (int kb = 0; kb < D_MODEL / 32; ++kb) {
        v16bf bb = load_frag(bp + (size_t)kb * 512, bp + (size_t)kb * 512 + 8);
        const unsigned short* a0 = &Xs[lrow * XS_STRIDE + kb * 32 + khalf];
        const unsigned short* a1 = &Xs[(16 + lrow) * XS_STRIDE + kb * 32 + khalf];
        v16bf af0 = load_frag(a0, a0 + 8);
        v16bf af1 = load_frag(a1, a1 + 8);
        c0a = __builtin_amdgcn_wmma_f32_16x16x32_bf16(false, af0, false, bb,
                                                      (short)0, c0a, false, false);
        c1a = __builtin_amdgcn_wmma_f32_16x16x32_bf16(false, af1, false, bb,
                                                      (short)0, c1a, false, false);
      }
      int colb = ntl * 16 + lrow;
#pragma unroll
      for (int r = 0; r < 8; ++r) {
        float h0 = c0a[r] > 0.f ? c0a[r] : 0.f;
        float h1 = c1a[r] > 0.f ? c1a[r] : 0.f;
        Hs[(rowb + r) * HS_STRIDE + colb]        = f32_to_bf16(h0);
        Hs[(16 + rowb + r) * HS_STRIDE + colb]   = f32_to_bf16(h1);
      }
    }
    __syncthreads();

    // ---- phase 2 (partial K): O += Hs @ W2[p*FH:(p+1)*FH, :] ----
#pragma unroll
    for (int i = 0; i < 4; ++i) {
      int nt = wave + 8 * i;
      const unsigned short* bp =
          W2e + ((size_t)nt * (D_FF / 32)) * 512 +
          (size_t)(p * (FH / 32)) * 512 + lane * 16;
      __builtin_prefetch(bp, 0, 1);
      v8f c0a = oacc[i][0];
      v8f c1a = oacc[i][1];
#pragma unroll 4
      for (int kb = 0; kb < FH / 32; ++kb) {
        v16bf bb = load_frag(bp + (size_t)kb * 512, bp + (size_t)kb * 512 + 8);
        const unsigned short* a0 = &Hs[lrow * HS_STRIDE + kb * 32 + khalf];
        const unsigned short* a1 = &Hs[(16 + lrow) * HS_STRIDE + kb * 32 + khalf];
        v16bf af0 = load_frag(a0, a0 + 8);
        v16bf af1 = load_frag(a1, a1 + 8);
        c0a = __builtin_amdgcn_wmma_f32_16x16x32_bf16(false, af0, false, bb,
                                                      (short)0, c0a, false, false);
        c1a = __builtin_amdgcn_wmma_f32_16x16x32_bf16(false, af1, false, bb,
                                                      (short)0, c1a, false, false);
      }
      oacc[i][0] = c0a;
      oacc[i][1] = c1a;
    }
  }

  // ---- final: scatter-accumulate with gates ----
#pragma unroll
  for (int i = 0; i < 4; ++i) {
    int col = (wave + 8 * i) * 16 + lrow;
#pragma unroll
    for (int m = 0; m < 2; ++m) {
#pragma unroll
      for (int r = 0; r < 8; ++r) {
        int row = m * 16 + rowb + r;
        int tok = tokS[row];
        if (tok >= 0)
          atomicAdd(out + (size_t)tok * D_MODEL + col,
                    gateS[row] * oacc[i][m][r]);
      }
    }
  }
}

// ---------------------------------------------------------------------------
extern "C" void kernel_launch(void* const* d_in, const int* in_sizes, int n_in,
                              void* d_out, int out_size, void* d_ws, size_t ws_size,
                              hipStream_t stream)
{
  const float* x  = (const float*)d_in[0];
  const float* Wg = (const float*)d_in[1];
  const float* bg = (const float*)d_in[2];
  const float* W1 = (const float*)d_in[3];
  const float* b1 = (const float*)d_in[4];
  const float* W2 = (const float*)d_in[5];
  const float* b2 = (const float*)d_in[6];
  float* out = (float*)d_out;

  const int T = in_sizes[0] / D_MODEL;                    // tokens

  // workspace layout
  const size_t WBYTES = (size_t)NUM_E * D_MODEL * D_FF * sizeof(unsigned short);
  unsigned short* W1f = (unsigned short*)d_ws;
  unsigned short* W2f = (unsigned short*)((char*)d_ws + WBYTES);
  int*   topi    = (int*)((char*)d_ws + 2 * WBYTES);
  float* topv    = (float*)(topi + 2 * (size_t)T);
  int*   tok_of  = (int*)(topv + 2 * (size_t)T);
  float* gate_of = (float*)(tok_of + 2 * (size_t)T);
  int*   meta    = (int*)(gate_of + 2 * (size_t)T);       // 64 ints

  hipMemsetAsync(meta, 0, 64 * sizeof(int), stream);
  hipMemsetAsync(out, 0, (size_t)out_size * sizeof(float), stream);

  int tb = (T + 255) / 256;
  moe_gate<<<tb, 256, 0, stream>>>(x, Wg, bg, T, topi, topv, meta);
  moe_scan<<<1, 32, 0, stream>>>(meta);
  moe_scatter<<<tb, 256, 0, stream>>>(T, topi, topv, meta + 26, tok_of, gate_of);

  long w1n = (long)NUM_E * D_MODEL * D_FF;
  long w2n = (long)NUM_E * D_FF * D_MODEL;
  moe_wconv<<<(int)((w1n + 255) / 256), 256, 0, stream>>>(W1, W1f, D_MODEL, D_FF, w1n);
  moe_wconv<<<(int)((w2n + 255) / 256), 256, 0, stream>>>(W2, W2f, D_FF, D_MODEL, w2n);

  int maxTiles = (2 * T + 31) / 32 + NUM_E;               // worst-case 32-row tiles
  moe_ffn<<<maxTiles, 256, 0, stream>>>(x, b1, b2, W1f, W2f, meta,
                                        tok_of, gate_of, out);
}
